// ODEFunc_46806553592570
// MI455X (gfx1250) — matrix operand order repaired and checked by reading, not compiled, not run on hardware
//
#include <hip/hip_runtime.h>
#include <hip/hip_bf16.h>
#include <math.h>

typedef __attribute__((ext_vector_type(16))) _Float16 v16h;
typedef __attribute__((ext_vector_type(8)))  float    v8f;

#define NN 8192
#define CW 64   // D*H2 = 4*16 feature width of src/tgt
#define AGG_WAVES 16      // waves per aggregation workgroup
#define KCHUNK (NN / AGG_WAVES)   // 512 rows of adj per wave
#define KITERS (KCHUNK / 32)      // 16 WMMA K-steps per wave

// -------- kernel 1a: g-MLP (4 -> 64 -> 64 -> 4), one thread per node --------
__global__ void __launch_bounds__(128)
k_gmlp(const float* __restrict__ x,
       const float* __restrict__ gW1, const float* __restrict__ gb1,
       const float* __restrict__ gW2, const float* __restrict__ gb2,
       const float* __restrict__ gW3, const float* __restrict__ gb3,
       float* __restrict__ out)
{
    const int n = blockIdx.x * blockDim.x + threadIdx.x;
    if (n >= NN) return;
    float xv[4];
    #pragma unroll
    for (int d = 0; d < 4; ++d) xv[d] = x[n * 4 + d];

    float h1[64];
    #pragma unroll
    for (int j = 0; j < 64; ++j) {
        float s = gb1[j];
        #pragma unroll
        for (int d = 0; d < 4; ++d) s += xv[d] * gW1[d * 64 + j];
        h1[j] = tanhf(s);
    }
    float h2[64];
    #pragma unroll
    for (int j = 0; j < 64; ++j) {
        float s = gb2[j];
        #pragma unroll
        for (int k = 0; k < 64; ++k) s += h1[k] * gW2[k * 64 + j];
        h2[j] = tanhf(s);
    }
    #pragma unroll
    for (int d = 0; d < 4; ++d) {
        float s = gb3[d];
        #pragma unroll
        for (int k = 0; k < 64; ++k) s += h2[k] * gW3[k * 4 + d];
        out[n * 4 + d] = s;
    }
}

// -------- kernel 1b: src (1->16->16->16) and tgt (1->16->16) MLPs -----------
// One thread per (node, d). Writes srcT transposed [64][N] in f16 (so WMMA
// B-fragments are contiguous v16h loads) and tgt row-major [N][64] in f32.
__global__ void __launch_bounds__(256)
k_srctgt(const float* __restrict__ x,
         const float* __restrict__ sW1, const float* __restrict__ sb1,
         const float* __restrict__ sW2, const float* __restrict__ sb2,
         const float* __restrict__ sW3, const float* __restrict__ sb3,
         const float* __restrict__ tW1, const float* __restrict__ tb1,
         const float* __restrict__ tW2, const float* __restrict__ tb2,
         _Float16* __restrict__ srcT, float* __restrict__ tgt)
{
    const int tid = blockIdx.x * blockDim.x + threadIdx.x;   // 0..4*NN-1
    const int d = tid >> 13;           // same d within a wave
    const int n = tid & (NN - 1);      // consecutive n across lanes -> coalesced
    const float u = x[n * 4 + d];

    float a1[16], a2[16];
    #pragma unroll
    for (int h = 0; h < 16; ++h) a1[h] = tanhf(u * sW1[h] + sb1[h]);
    #pragma unroll
    for (int h = 0; h < 16; ++h) {
        float s = sb2[h];
        #pragma unroll
        for (int k = 0; k < 16; ++k) s += a1[k] * sW2[k * 16 + h];
        a2[h] = tanhf(s);
    }
    #pragma unroll
    for (int h = 0; h < 16; ++h) {
        float s = sb3[h];
        #pragma unroll
        for (int k = 0; k < 16; ++k) s += a2[k] * sW3[k * 16 + h];
        srcT[(size_t)(d * 16 + h) * NN + n] = (_Float16)s;
    }

    float b1[16];
    #pragma unroll
    for (int h = 0; h < 16; ++h) b1[h] = tanhf(u * tW1[h] + tb1[h]);
    #pragma unroll
    for (int h = 0; h < 16; ++h) {
        float s = tb2[h];
        #pragma unroll
        for (int k = 0; k < 16; ++k) s += b1[k] * tW2[k * 16 + h];
        tgt[(size_t)n * CW + d * 16 + h] = s;
    }
}

// -------- kernel 2: S = adj^T @ src via v_wmma_f32_16x16x32_f16, then -------
// interaction[j,d] = sum_h S[j,d*16+h] * tgt[j,d*16+h], added onto out.
// One workgroup per 16-column j-tile; 16 waves split K=8192 into chunks of 512.
// adj is a 256 MB once-only stream: loads are software-pipelined (next
// iteration prefetched into registers) and marked non-temporal so they don't
// evict the hot 1 MB srcT / 2 MB tgt working set from the 192 MB L2.
__global__ void __launch_bounds__(32 * AGG_WAVES)
k_aggregate(const float* __restrict__ adj,
            const _Float16* __restrict__ srcT,
            const float* __restrict__ tgt,
            float* __restrict__ out)
{
    __shared__ float red[AGG_WAVES][4][8][32];   // 64 KB

    const int j0   = blockIdx.x * 16;
    const int tid  = threadIdx.x;
    const int wave = tid >> 5;
    const int lane = tid & 31;
    const int m    = lane & 15;   // A-fragment row (j), B-fragment col (c)
    const int hi   = lane >> 4;   // lane-group select

    v8f acc[4];
    #pragma unroll
    for (int t = 0; t < 4; ++t)
        #pragma unroll
        for (int r = 0; r < 8; ++r) acc[t][r] = 0.0f;

    // A-fragment base: column (j0+m) of adj, row offset 16*hi.
    // Contiguous A layout (matches B): half q of lane L holds
    //   K = q + 16*(L>=16),  M = L&15.
    const float* aptr = adj + (size_t)(16 * hi) * NN + (size_t)(j0 + m);
    const int kbase = wave * KCHUNK;

    // prime the pipeline: adj rows for k-step 0 (non-temporal: single use)
    float af[16];
    #pragma unroll
    for (int q = 0; q < 16; ++q)
        af[q] = __builtin_nontemporal_load(
                    aptr + (size_t)(kbase + q) * NN);

    for (int it = 0; it < KITERS; ++it) {
        const int k = kbase + it * 32;

        v16h a;
        #pragma unroll
        for (int q = 0; q < 16; ++q) a[q] = (_Float16)af[q];

        // prefetch next k-step (last iteration re-reads kbase: in-bounds, unused)
        const int knext = (it == KITERS - 1) ? kbase : (k + 32);
        #pragma unroll
        for (int q = 0; q < 16; ++q)
            af[q] = __builtin_nontemporal_load(
                        aptr + (size_t)(knext + q) * NN);

        #pragma unroll
        for (int t = 0; t < 4; ++t) {
            // B tile = src[k..k+31, 16t..16t+15]; contiguous order, same as A:
            // half q of lane L holds K = q + 16*(L>=16), N = L&15.
            // srcT is transposed -> one contiguous 32-byte v16h per lane (L2-hit).
            const v16h b = *(const v16h*)(srcT + (size_t)(t * 16 + m) * NN
                                               + (size_t)(k + 16 * hi));
            acc[t] = __builtin_amdgcn_wmma_f32_16x16x32_f16(
                         false, a, false, b, (short)0, acc[t], false, false);
        }
    }

    #pragma unroll
    for (int t = 0; t < 4; ++t)
        #pragma unroll
        for (int r = 0; r < 8; ++r)
            red[wave][t][r][lane] = acc[t][r];
    __syncthreads();

    // Waves 0..3 each finish one c-tile t (t == d since c = d*16+h).
    if (wave < 4) {
        const int t = wave;
        #pragma unroll
        for (int r = 0; r < 8; ++r) {
            float s = 0.0f;
            #pragma unroll
            for (int w2 = 0; w2 < AGG_WAVES; ++w2) s += red[w2][t][r][lane];
            // D layout: VGPR r, lanes 0-15 -> M=r, lanes 16-31 -> M=r+8; N=lane&15
            const int j = j0 + r + 8 * hi;
            float v = s * tgt[(size_t)j * CW + t * 16 + m];
            v += __shfl_xor(v, 1);
            v += __shfl_xor(v, 2);
            v += __shfl_xor(v, 4);
            v += __shfl_xor(v, 8);
            if (m == 0) out[j * 4 + t] += v;   // single writer per (j,t)
        }
    }
}

extern "C" void kernel_launch(void* const* d_in, const int* in_sizes, int n_in,
                              void* d_out, int out_size, void* d_ws, size_t ws_size,
                              hipStream_t stream)
{
    const float* x   = (const float*)d_in[0];
    const float* adj = (const float*)d_in[1];
    const float* gW1 = (const float*)d_in[2];
    const float* gb1 = (const float*)d_in[3];
    const float* gW2 = (const float*)d_in[4];
    const float* gb2 = (const float*)d_in[5];
    const float* gW3 = (const float*)d_in[6];
    const float* gb3 = (const float*)d_in[7];
    const float* sW1 = (const float*)d_in[8];
    const float* sb1 = (const float*)d_in[9];
    const float* sW2 = (const float*)d_in[10];
    const float* sb2 = (const float*)d_in[11];
    const float* sW3 = (const float*)d_in[12];
    const float* sb3 = (const float*)d_in[13];
    const float* tW1 = (const float*)d_in[14];
    const float* tb1 = (const float*)d_in[15];
    const float* tW2 = (const float*)d_in[16];
    const float* tb2 = (const float*)d_in[17];

    float* out = (float*)d_out;
    _Float16* srcT = (_Float16*)d_ws;                                        // 1 MB
    float* tgt = (float*)((char*)d_ws + (size_t)CW * NN * sizeof(_Float16)); // 2 MB

    k_gmlp<<<NN / 128, 128, 0, stream>>>(x, gW1, gb1, gW2, gb2, gW3, gb3, out);
    k_srctgt<<<(NN * 4) / 256, 256, 0, stream>>>(x, sW1, sb1, sW2, sb2, sW3, sb3,
                                                 tW1, tb1, tW2, tb2, srcT, tgt);
    k_aggregate<<<NN / 16, 32 * AGG_WAVES, 0, stream>>>(adj, srcT, tgt, out);
}
